// MultiHeadAttention_77730318123153
// MI455X (gfx1250) — matrix operand order repaired and checked by reading, not compiled, not run on hardware
//
#include <hip/hip_runtime.h>
#include <hip/hip_bf16.h>

// MHA forward for MI455X (gfx1250): all GEMMs via v_wmma_f32_16x16x32_bf16,
// LDS staging via gfx1250 async-to/from-LDS ops where available.
// B=32 L=1024 D=256 H=4 DH=128.  Outputs: out[B,L,D] fp32 then attn[B,H,L,L] fp32.

#define B_   32
#define L_   1024
#define D_   256
#define H_   4
#define DH_  128
#define NQK  512      // H*DH

typedef __bf16 bf16_t;
typedef __attribute__((ext_vector_type(16))) __bf16 v16bf;
typedef __attribute__((ext_vector_type(8)))  float  v8f;
typedef int v4i __attribute__((vector_size(4 * sizeof(int))));

#define AS1 __attribute__((address_space(1)))
#define AS3 __attribute__((address_space(3)))

#if defined(__gfx1250__) && \
    __has_builtin(__builtin_amdgcn_global_load_async_to_lds_b128) && \
    __has_builtin(__builtin_amdgcn_global_store_async_from_lds_b128) && \
    __has_builtin(__builtin_amdgcn_s_wait_asynccnt)
#define HAVE_ASYNC_LDS 1
#else
#define HAVE_ASYNC_LDS 0
#endif

#if HAVE_ASYNC_LDS
static __device__ __forceinline__ void async_g2l_b128(const void* g, void* l) {
  __builtin_amdgcn_global_load_async_to_lds_b128((AS1 v4i*)g, (AS3 v4i*)l, 0, 0);
}
static __device__ __forceinline__ void async_l2g_b128(void* g, const void* l) {
  __builtin_amdgcn_global_store_async_from_lds_b128((AS1 v4i*)g, (AS3 v4i*)l, 0, 0);
}
#endif

// -------- fragment helpers (CDNA5 16-bit WMMA VGPR layouts, ISA 7.12.2) -----
static __device__ __forceinline__ v16bf load_frag2(const bf16_t* c0, const bf16_t* c1) {
  union { v16bf v; uint4 u[2]; } r;
  r.u[0] = *(const uint4*)c0;   // 8 bf16
  r.u[1] = *(const uint4*)c1;   // 8 bf16
  return r.v;
}

#define ASTR 40   // 32 bf16 + 8 pad: 80B row stride, 16B aligned, kills bank conflicts

// ============================================================================
// Kernel 1: projection GEMM  out = X @ W^T + bias  (fp32 in, bf16 out)
//   X:[B,L,256]  W:[512,256]  grid=(L/128, H, B), 256 thr = 8 waves.
//   transposed==0: out bf16 [B*H, L, DH]   (Q, K)
//   transposed==1: out bf16 [B*H, DH, L]   (V, pre-transposed for P@V B-frags)
// ============================================================================
__global__ __launch_bounds__(256)
void qkv_proj_kernel(const float* __restrict__ X, const float* __restrict__ W,
                     const float* __restrict__ bias, bf16_t* __restrict__ out,
                     int transposed)
{
  __shared__ bf16_t As[128 * ASTR];
  __shared__ bf16_t Bs[128 * ASTR];
  const int b  = blockIdx.z;
  const int m0 = blockIdx.x * 128;           // row (l) base
  const int h  = blockIdx.y;                 // N/128 == H
  const int n0 = h * 128;                    // output-feature base
  const int t  = threadIdx.x;
  const int w  = t >> 5, lane = t & 31;
  const int lm = lane & 15, hi = lane >> 4;

  v8f acc[8];
  const v8f vzero = {0.f,0.f,0.f,0.f,0.f,0.f,0.f,0.f};
#pragma unroll
  for (int i = 0; i < 8; ++i) acc[i] = vzero;

  const float* Xb = X + ((size_t)b * L_ + m0) * D_;

  for (int kb = 0; kb < D_ / 32; ++kb) {     // 8 K-steps
    const int k0 = kb * 32;
#pragma unroll
    for (int j = 0; j < 4; ++j) {            // stage A and B tiles, fp32 -> bf16
      const int f = t + j * 256;             // float4 id, 1024 per tile
      const int row = f >> 3, c4 = (f & 7) * 4;
      float4 xa = *(const float4*)(Xb + (size_t)row * D_ + k0 + c4);
      union { bf16_t hh[4]; uint2 u; } pa;
      pa.hh[0]=(bf16_t)xa.x; pa.hh[1]=(bf16_t)xa.y; pa.hh[2]=(bf16_t)xa.z; pa.hh[3]=(bf16_t)xa.w;
      *(uint2*)&As[row * ASTR + c4] = pa.u;
      float4 wb = *(const float4*)(W + (size_t)(n0 + row) * D_ + k0 + c4);
      union { bf16_t hh[4]; uint2 u; } pb;
      pb.hh[0]=(bf16_t)wb.x; pb.hh[1]=(bf16_t)wb.y; pb.hh[2]=(bf16_t)wb.z; pb.hh[3]=(bf16_t)wb.w;
      *(uint2*)&Bs[row * ASTR + c4] = pb.u;
      if (kb + 1 < D_ / 32) {                // prefetch next K tile (global_prefetch_b8)
        __builtin_prefetch(Xb + (size_t)row * D_ + k0 + 32 + c4, 0, 3);
        __builtin_prefetch(W + (size_t)(n0 + row) * D_ + k0 + 32 + c4, 0, 3);
      }
    }
    __syncthreads();

    const int am   = w * 16 + lm;            // wave's 16-row strip
    const int ksel = hi * 8;
    v16bf afrag = load_frag2(&As[am * ASTR + ksel], &As[am * ASTR + ksel + 16]);
#pragma unroll
    for (int nt = 0; nt < 8; ++nt) {
      const int bn = nt * 16 + lm;
      const int kh = hi * 8;
      v16bf bfrag = load_frag2(&Bs[bn * ASTR + kh], &Bs[bn * ASTR + kh + 16]);
      acc[nt] = __builtin_amdgcn_wmma_f32_16x16x32_bf16(
          false, afrag, false, bfrag, (short)0, acc[nt], false, false);
    }
    __syncthreads();
  }

#pragma unroll
  for (int nt = 0; nt < 8; ++nt) {
    const int n  = n0 + nt * 16 + lm;
    const int dh = n & (DH_ - 1);
    const float bv = bias[n];
    if (!transposed) {
#pragma unroll
      for (int r = 0; r < 8; ++r) {
        const int l = m0 + w * 16 + r + hi * 8;
        out[(((size_t)b * H_ + h) * L_ + l) * DH_ + dh] = (bf16_t)(acc[nt][r] + bv);
      }
    } else {                                 // [BH, DH, L]: 8 consecutive l -> one b128
      union { bf16_t hh[8]; uint4 u; } pk;
#pragma unroll
      for (int r = 0; r < 8; ++r) pk.hh[r] = (bf16_t)(acc[nt][r] + bv);
      const int l = m0 + w * 16 + hi * 8;
      *(uint4*)&out[(((size_t)b * H_ + h) * DH_ + dh) * L_ + l] = pk.u;
    }
  }
}

// ============================================================================
// Kernel 2: fused attention. One block = 64 query rows of one (b,h).
//   S = Q K^T (WMMA) -> full 64x1024 fp32 stripe in 257KB dynamic LDS
//   softmax in LDS -> async-store attn stripe from LDS overlapped with P@V
//   ctx = P V (WMMA, P fp32->bf16 from LDS, V read transposed)
// ============================================================================
#define SSTR 1028   // 1024 + 4: rows land on distinct bank groups for b128 reads
#define QSTR 136    // 128 + 8

__global__ __launch_bounds__(256)
void attn_kernel(const bf16_t* __restrict__ qws, const bf16_t* __restrict__ kws,
                 const bf16_t* __restrict__ vtws, float* __restrict__ attn_out,
                 bf16_t* __restrict__ ctxws)
{
  extern __shared__ float S[];               // [64][SSTR] fp32 scores / probs
  __shared__ bf16_t Qs[64 * QSTR];
  __shared__ float redbuf[64 * 4];
  __shared__ float rowmax[64];
  __shared__ float rowinv[64];

  const int bh = blockIdx.y;
  const int q0 = blockIdx.x * 64;
  const int t = threadIdx.x, w = t >> 5, lane = t & 31;
  const int lm = lane & 15, hi = lane >> 4;

  // ---- stage Q rows [q0, q0+64) x DH into LDS (bf16, 16B chunks) ----
  const bf16_t* Qg = qws + ((size_t)bh * L_ + q0) * DH_;
#pragma unroll
  for (int j = 0; j < 4; ++j) {
    const int f = t + j * 256;               // 8-elem chunk id, 1024 total
    const int row = f >> 4, c8 = (f & 15) * 8;
#if HAVE_ASYNC_LDS
    async_g2l_b128(Qg + (size_t)row * DH_ + c8, &Qs[row * QSTR + c8]);
#else
    *(uint4*)&Qs[row * QSTR + c8] = *(const uint4*)(Qg + (size_t)row * DH_ + c8);
#endif
  }
#if HAVE_ASYNC_LDS
  __builtin_amdgcn_s_wait_asynccnt(0);
#endif
  __syncthreads();

  // ---- S = Q K^T ----
  const int mt = w & 3;                      // wave's 16-row m-tile
  v16bf qf[4];                               // preload Q frags for all 4 K-steps of DH
#pragma unroll
  for (int kb = 0; kb < 4; ++kb) {
    const int base = mt * 16 + lm;
    const int ksel = kb * 32 + hi * 8;
    qf[kb] = load_frag2(&Qs[base * QSTR + ksel], &Qs[base * QSTR + ksel + 16]);
  }
  const bf16_t* Kg = kws + (size_t)bh * L_ * DH_;
  for (int nt = (w >> 2); nt < 64; nt += 2) {     // 32 key tiles per wave
    v8f acc = {0.f,0.f,0.f,0.f,0.f,0.f,0.f,0.f};
    const int kp = nt * 16 + lm;                  // key position for B-frag lane
#pragma unroll
    for (int kb = 0; kb < 4; ++kb) {
      const int dh0 = kb * 32 + hi * 8;
      const bf16_t* Kr = Kg + (size_t)kp * DH_ + dh0;   // two 16B chunks per lane
      v16bf bfrag = load_frag2(Kr, Kr + 16);
      acc = __builtin_amdgcn_wmma_f32_16x16x32_bf16(
          false, qf[kb], false, bfrag, (short)0, acc, false, false);
    }
#pragma unroll
    for (int r = 0; r < 8; ++r)
      S[(mt * 16 + r + hi * 8) * SSTR + nt * 16 + lm] = acc[r];
  }
  __syncthreads();

  // ---- softmax over each row of S (64 rows, 4 threads per row) ----
  {
    const int row = t >> 2, part = t & 3;
    const float* Sr = &S[row * SSTR + part * 256];
    float m = -__builtin_inff();
    for (int i = 0; i < 256; ++i) m = fmaxf(m, Sr[i]);
    redbuf[row * 4 + part] = m;
  }
  __syncthreads();
  if (t < 64)
    rowmax[t] = fmaxf(fmaxf(redbuf[t*4], redbuf[t*4+1]),
                      fmaxf(redbuf[t*4+2], redbuf[t*4+3]));
  __syncthreads();
  {
    const int row = t >> 2, part = t & 3;
    const float m = rowmax[row];
    const float* Sr = &S[row * SSTR + part * 256];
    float s = 0.f;
    for (int i = 0; i < 256; ++i) s += __expf(Sr[i] - m);
    redbuf[row * 4 + part] = s;
  }
  __syncthreads();
  if (t < 64)
    rowinv[t] = 1.0f / (redbuf[t*4] + redbuf[t*4+1] + redbuf[t*4+2] + redbuf[t*4+3]);
  __syncthreads();

  // ---- normalize P in LDS, then push the attn stripe to HBM ----
  float* Ao = attn_out + (size_t)bh * L_ * L_ + (size_t)q0 * L_;
#if HAVE_ASYNC_LDS
  for (int j = 0; j < 256; ++j) {
    const int idx = t + j * 256;             // 65536 probs per block
    const int row = idx >> 10, col = idx & 1023;
    S[row * SSTR + col] = __expf(S[row * SSTR + col] - rowmax[row]) * rowinv[row];
  }
  __syncthreads();
  // async LDS->HBM store of the whole 256KB stripe; overlaps with the P@V
  // WMMA loop below (S is read-only from here on; S_ENDPGM waits ASYNCcnt).
  for (int j = 0; j < 64; ++j) {
    const int c = t + j * 256;               // 16384 b128 chunks
    const int row = c >> 8, col = (c & 255) * 4;
    async_l2g_b128(Ao + (size_t)row * L_ + col, &S[row * SSTR + col]);
  }
#else
  for (int j = 0; j < 256; ++j) {
    const int idx = t + j * 256;
    const int row = idx >> 10, col = idx & 1023;
    const float p = __expf(S[row * SSTR + col] - rowmax[row]) * rowinv[row];
    S[row * SSTR + col] = p;
    Ao[(size_t)row * L_ + col] = p;
  }
  __syncthreads();
#endif

  // ---- ctx = P @ V  (V stored [BH, DH, L] so B-frags are contiguous) ----
  const bf16_t* Vg = vtws + (size_t)bh * DH_ * L_;
  const int ntb = (w >> 2) * 4;              // 4 n-tiles (dh) per wave
  v8f cacc[4];
  const v8f vzero = {0.f,0.f,0.f,0.f,0.f,0.f,0.f,0.f};
#pragma unroll
  for (int i = 0; i < 4; ++i) cacc[i] = vzero;

  for (int kb = 0; kb < 32; ++kb) {          // K = 1024 keys, 32 per WMMA
    const int k0 = kb * 32;
    v16bf af;                                // P fragment: fp32 LDS -> bf16
    {
      const float* Pr = &S[(mt * 16 + lm) * SSTR + k0 + hi * 8];
      float4 c0 = *(const float4*)(Pr);
      float4 c1 = *(const float4*)(Pr + 4);
      float4 c2 = *(const float4*)(Pr + 16);
      float4 c3 = *(const float4*)(Pr + 20);
      af[0]=(bf16_t)c0.x;  af[1]=(bf16_t)c0.y;  af[2]=(bf16_t)c0.z;  af[3]=(bf16_t)c0.w;
      af[4]=(bf16_t)c1.x;  af[5]=(bf16_t)c1.y;  af[6]=(bf16_t)c1.z;  af[7]=(bf16_t)c1.w;
      af[8]=(bf16_t)c2.x;  af[9]=(bf16_t)c2.y;  af[10]=(bf16_t)c2.z; af[11]=(bf16_t)c2.w;
      af[12]=(bf16_t)c3.x; af[13]=(bf16_t)c3.y; af[14]=(bf16_t)c3.z; af[15]=(bf16_t)c3.w;
    }
#pragma unroll
    for (int i = 0; i < 4; ++i) {
      const int dh = (ntb + i) * 16 + lm;
      const bf16_t* Vr = Vg + (size_t)dh * L_ + k0 + hi * 8;  // two 16B chunks
      v16bf bfrag = load_frag2(Vr, Vr + 16);
      cacc[i] = __builtin_amdgcn_wmma_f32_16x16x32_bf16(
          false, af, false, bfrag, (short)0, cacc[i], false, false);
    }
  }

  // store ctx as bf16 [B, L, H*DH] for the output projection
  const int b = bh >> 2, h = bh & 3;
#pragma unroll
  for (int i = 0; i < 4; ++i) {
    const int dh = (ntb + i) * 16 + lm;
#pragma unroll
    for (int r = 0; r < 8; ++r) {
      const int l = q0 + mt * 16 + r + hi * 8;
      ctxws[((size_t)b * L_ + l) * NQK + h * DH_ + dh] = (bf16_t)cacc[i][r];
    }
  }
}

// ============================================================================
// Kernel 3: out = ctx @ Wf^T + bf   (ctx bf16 [B*L,512], Wf fp32 [256,512])
//   grid=(B*L/128, D/128), fp32 output with bias epilogue.
// ============================================================================
__global__ __launch_bounds__(256)
void out_proj_kernel(const bf16_t* __restrict__ Ctx, const float* __restrict__ Wf,
                     const float* __restrict__ bfv, float* __restrict__ out)
{
  __shared__ bf16_t As[128 * ASTR];
  __shared__ bf16_t Bs[128 * ASTR];
  const int m0 = blockIdx.x * 128;
  const int n0 = blockIdx.y * 128;
  const int t = threadIdx.x, w = t >> 5, lane = t & 31;
  const int lm = lane & 15, hi = lane >> 4;

  v8f acc[8];
  const v8f vzero = {0.f,0.f,0.f,0.f,0.f,0.f,0.f,0.f};
#pragma unroll
  for (int i = 0; i < 8; ++i) acc[i] = vzero;

  for (int kb = 0; kb < NQK / 32; ++kb) {    // 16 K-steps
    const int k0 = kb * 32;
#pragma unroll
    for (int j = 0; j < 2; ++j) {            // stage A: bf16, pure copy -> async
      const int f = t + j * 256;             // 8-chunk id, 512 total
      const int row = f >> 2, c8 = (f & 3) * 8;
#if HAVE_ASYNC_LDS
      async_g2l_b128(Ctx + (size_t)(m0 + row) * NQK + k0 + c8, &As[row * ASTR + c8]);
#else
      *(uint4*)&As[row * ASTR + c8] =
          *(const uint4*)(Ctx + (size_t)(m0 + row) * NQK + k0 + c8);
#endif
    }
#pragma unroll
    for (int j = 0; j < 4; ++j) {            // stage B: Wf fp32 -> bf16
      const int f = t + j * 256;
      const int row = f >> 3, c4 = (f & 7) * 4;
      float4 wv = *(const float4*)(Wf + (size_t)(n0 + row) * NQK + k0 + c4);
      union { bf16_t hh[4]; uint2 u; } pb;
      pb.hh[0]=(bf16_t)wv.x; pb.hh[1]=(bf16_t)wv.y; pb.hh[2]=(bf16_t)wv.z; pb.hh[3]=(bf16_t)wv.w;
      *(uint2*)&Bs[row * ASTR + c4] = pb.u;
    }
#if HAVE_ASYNC_LDS
    __builtin_amdgcn_s_wait_asynccnt(0);
#endif
    __syncthreads();

    const int am   = w * 16 + lm;
    const int ksel = hi * 8;
    v16bf afrag = load_frag2(&As[am * ASTR + ksel], &As[am * ASTR + ksel + 16]);
#pragma unroll
    for (int nt = 0; nt < 8; ++nt) {
      const int bn = nt * 16 + lm;
      const int kh = hi * 8;
      v16bf bfrag = load_frag2(&Bs[bn * ASTR + kh], &Bs[bn * ASTR + kh + 16]);
      acc[nt] = __builtin_amdgcn_wmma_f32_16x16x32_bf16(
          false, afrag, false, bfrag, (short)0, acc[nt], false, false);
    }
    __syncthreads();
  }

#pragma unroll
  for (int nt = 0; nt < 8; ++nt) {
    const int n = n0 + nt * 16 + lm;
    const float bv = bfv[n];
#pragma unroll
    for (int r = 0; r < 8; ++r) {
      const int m = m0 + w * 16 + r + hi * 8;
      out[(size_t)m * D_ + n] = acc[nt][r] + bv;
    }
  }
}

// ============================================================================
extern "C" void kernel_launch(void* const* d_in, const int* in_sizes, int n_in,
                              void* d_out, int out_size, void* d_ws, size_t ws_size,
                              hipStream_t stream) {
  (void)in_sizes; (void)n_in; (void)out_size; (void)ws_size;
  const float* key   = (const float*)d_in[0];
  const float* value = (const float*)d_in[1];
  const float* query = (const float*)d_in[2];
  const float* Wq = (const float*)d_in[3];
  const float* bq = (const float*)d_in[4];
  const float* Wk = (const float*)d_in[5];
  const float* bk = (const float*)d_in[6];
  const float* Wv = (const float*)d_in[7];
  const float* bv = (const float*)d_in[8];
  const float* Wf = (const float*)d_in[9];
  const float* bf = (const float*)d_in[10];

  float* out  = (float*)d_out;                          // [B,L,D]
  float* attn = out + (size_t)B_ * L_ * D_;             // [B,H,L,L]

  // workspace: Q bf16 [BH,L,DH] | K bf16 [BH,L,DH] | V^T bf16 [BH,DH,L] | ctx bf16 [B,L,512]
  const size_t qkv_elems = (size_t)B_ * H_ * L_ * DH_;  // 16,777,216 each
  bf16_t* qws   = (bf16_t*)d_ws;
  bf16_t* kws   = qws + qkv_elems;
  bf16_t* vtws  = kws + qkv_elems;
  bf16_t* ctxws = vtws + qkv_elems;

  const dim3 blk(256);

  // 1) projections (bf16 WMMA); V written pre-transposed
  const dim3 g1(L_ / 128, H_, B_);
  qkv_proj_kernel<<<g1, blk, 0, stream>>>(query, Wq, bq, qws, 0);
  qkv_proj_kernel<<<g1, blk, 0, stream>>>(key,   Wk, bk, kws, 0);
  qkv_proj_kernel<<<g1, blk, 0, stream>>>(value, Wv, bv, vtws, 1);

  // 2) fused attention: 257KB dynamic LDS score stripe (CDNA5 320KB/WG LDS)
  const size_t smem = (size_t)64 * SSTR * sizeof(float);
  (void)hipFuncSetAttribute(reinterpret_cast<const void*>(attn_kernel),
                            hipFuncAttributeMaxDynamicSharedMemorySize, (int)smem);
  attn_kernel<<<dim3(L_ / 64, B_ * H_), blk, smem, stream>>>(qws, kws, vtws, attn, ctxws);

  // 3) output projection
  out_proj_kernel<<<dim3((B_ * L_) / 128, D_ / 128), blk, 0, stream>>>(ctxws, Wf, bf, out);
}
